// Block_68633577390708
// MI455X (gfx1250) — matrix operand order repaired and checked by reading, not compiled, not run on hardware
//
#include <hip/hip_runtime.h>

// ---------------------------------------------------------------------------
// Transformer block for MI455X (gfx1250, wave32, WMMA).
// All GEMMs + attention run on v_wmma_f32_16x16x32_bf16 (f32 accumulate).
// Problem: B=2, T=2048, D=1024, H=16, HS=64.  ~137 GFLOP, ~0.1 GB traffic
// => compute bound; bf16 WMMA is the right precision/path on CDNA5.
// GEMM: 128x256 block tile, 8 waves of 64x64, K-step 32, double-buffered LDS
// filled with GLOBAL_LOAD_ASYNC_TO_LDS_B128 (ASYNCcnt) so the copy path uses
// no VGPRs and fully overlaps the 16 WMMAs per K-step; one barrier per step.
// ---------------------------------------------------------------------------

typedef __attribute__((ext_vector_type(16))) __bf16 v16bf;
typedef __attribute__((ext_vector_type(8)))  __bf16 v8bf;
typedef __attribute__((ext_vector_type(8)))  float  v8f;

typedef unsigned short u16;
typedef unsigned int   u32;

static constexpr int BATCH = 2;
static constexpr int SEQ   = 2048;
static constexpr int DIM   = 1024;
static constexpr int NHEAD = 16;
static constexpr int HEADSZ= 64;
static constexpr int MROWS = BATCH * SEQ;      // 4096
static constexpr int FFDIM = 4 * DIM;          // 4096
static constexpr int QKVN  = 3 * DIM;          // 3072 (fused QKV projection)

__device__ __forceinline__ u16 f2bf(float f) {
    u32 u = __float_as_uint(f);
    u32 r = u + 0x7FFFu + ((u >> 16) & 1u);    // round-to-nearest-even
    return (u16)(r >> 16);
}

__device__ __forceinline__ v8f wmma_bf16(v16bf a, v16bf b, v8f c) {
    // (neg_a, A, neg_b, B, c_mod, C, reuse_a, reuse_b)
    return __builtin_amdgcn_wmma_f32_16x16x32_bf16(false, a, false, b,
                                                   (short)0, c, false, false);
}

union Frag16 { v16bf v; v8bf h[2]; };

// Async memory->LDS copy, 16B per lane; offset applies to BOTH addresses
// (ISA 15.18.3); ASYNCcnt tracked.
template <int OFF>
__device__ __forceinline__ void async_ld_b128(u32 lds, const u16* gp) {
    asm volatile("global_load_async_to_lds_b128 %0, %1, off offset:%2"
                 :: "v"(lds), "v"(gp), "i"(OFF) : "memory");
}
__device__ __forceinline__ void wait_async0() {
    asm volatile("s_wait_asynccnt 0x0" ::: "memory");
}

// ---------------------------------------------------------------------------
// Weight repack: fp32 [K][N] -> bf16 [N][K] (B-transposed, K contiguous)
// ---------------------------------------------------------------------------
__global__ void transpose_conv_kernel(const float* __restrict__ w,
                                      u16* __restrict__ out, int K, int N) {
    int idx = blockIdx.x * 256 + threadIdx.x;
    if (idx >= K * N) return;
    int n = idx / K, k = idx - n * K;
    out[idx] = f2bf(w[(size_t)k * N + n]);
}

// Wq/Wk/Wv: fp32 [NHEAD][DIM][HEADSZ] -> bf16 [NHEAD*HEADSZ][DIM]
__global__ void repack_qkv_kernel(const float* __restrict__ w,
                                  u16* __restrict__ out) {
    int idx = blockIdx.x * 256 + threadIdx.x;
    if (idx >= DIM * DIM) return;
    int n = idx / DIM, k = idx - n * DIM;
    int hh = n >> 6, hs = n & 63;
    out[idx] = f2bf(w[((size_t)hh * DIM + k) * HEADSZ + hs]);
}

// ---------------------------------------------------------------------------
// LayerNorm: fp32 row -> bf16 row (256 threads per 1024-wide row)
// ---------------------------------------------------------------------------
__global__ __launch_bounds__(256) void ln_kernel(const float* __restrict__ x,
                                                 const float* __restrict__ g,
                                                 const float* __restrict__ bsh,
                                                 u16* __restrict__ out) {
    __shared__ float ssum[8], ssq[8], stats[2];
    const int row = blockIdx.x;
    const int tid = threadIdx.x;
    const float4 xv = *(const float4*)(x + (size_t)row * DIM + tid * 4);
    float s = xv.x + xv.y + xv.z + xv.w;
    float q = xv.x * xv.x + xv.y * xv.y + xv.z * xv.z + xv.w * xv.w;
    for (int m = 16; m >= 1; m >>= 1) { s += __shfl_xor(s, m); q += __shfl_xor(q, m); }
    if ((tid & 31) == 0) { ssum[tid >> 5] = s; ssq[tid >> 5] = q; }
    __syncthreads();
    if (tid == 0) {
        float S = 0.f, Q = 0.f;
        for (int i = 0; i < 8; i++) { S += ssum[i]; Q += ssq[i]; }
        float mu = S / DIM;
        float var = Q / DIM - mu * mu;
        stats[0] = mu; stats[1] = rsqrtf(var + 1e-5f);
    }
    __syncthreads();
    const float mu = stats[0], rs = stats[1];
    const int c = tid * 4;
    const float xi[4] = {xv.x, xv.y, xv.z, xv.w};
    u16 ob[4];
    for (int i = 0; i < 4; i++)
        ob[i] = f2bf((xi[i] - mu) * rs * g[c + i] + bsh[c + i]);
    *(uint2*)(out + (size_t)row * DIM + c) = *(const uint2*)ob;
}

// ---------------------------------------------------------------------------
// Generic bf16 GEMM:  C[M][N] = A[M][K] * Bt[N][K]^T  (+bias, relu, resid)
// 256 threads / 8 waves; block tile 128x256; wave tile 64x64; K-step 32;
// double-buffered LDS filled by async loads (one barrier per K-step).
// ---------------------------------------------------------------------------
static constexpr int LSTR = 48;  // LDS row stride (elements), keeps 16B align

__global__ __launch_bounds__(256) void gemm_bf16_kernel(
    const u16* __restrict__ A, int lda,
    const u16* __restrict__ Bt, int ldb,
    int M, int N, int K,
    const float* __restrict__ bias,
    const float* __restrict__ resid, int ldr,
    void* __restrict__ out, int ldo, int out_bf16, int relu)
{
    __shared__ u16 As[2][128 * LSTR];   // 2 x 12 KB
    __shared__ u16 Bs[2][256 * LSTR];   // 2 x 24 KB

    const int tid  = threadIdx.x;
    const int lane = tid & 31;
    const int wave = tid >> 5;
    const int wm = wave & 1;     // 2 waves along M (64 rows each)
    const int wn = wave >> 1;    // 4 waves along N (64 cols each)
    const int m0 = blockIdx.y * 128;
    const int n0 = blockIdx.x * 256;

    v8f acc[4][4] = {};

    const int sra = tid >> 1;          // A staging row 0..127
    const int ska = (tid & 1) * 16;    // A staging k offset 0/16

    // LDS byte addresses for this thread's async-copy destinations
    // (flat-aperture rule: low 32 bits of a shared pointer == LDS byte offset)
    const u32 ldsA[2] = { (u32)(uintptr_t)&As[0][sra * LSTR + ska],
                          (u32)(uintptr_t)&As[1][sra * LSTR + ska] };
    const u32 ldsB[2] = { (u32)(uintptr_t)&Bs[0][tid * LSTR],
                          (u32)(uintptr_t)&Bs[1][tid * LSTR] };

    auto stage = [&](int buf, int kk) {
        // A tile 128x32: 2 threads/row, 16 elems (32B) each
        const u16* ga = A + (size_t)(m0 + sra) * lda + kk + ska;
        async_ld_b128<0>(ldsA[buf], ga);
        async_ld_b128<16>(ldsA[buf], ga);
        // B tile 256x32: 1 thread/row, 32 elems (64B) each
        const u16* gb = Bt + (size_t)(n0 + tid) * ldb + kk;
        async_ld_b128<0>(ldsB[buf], gb);
        async_ld_b128<16>(ldsB[buf], gb);
        async_ld_b128<32>(ldsB[buf], gb);
        async_ld_b128<48>(ldsB[buf], gb);
    };

    stage(0, 0);
    wait_async0();
    __syncthreads();

    const int ak0 = (lane & 16) ? 8 : 0;    // A-frag K interleave per ISA
    const int bk0 = (lane & 16) ? 8 : 0;    // B-frag K interleave per ISA
    int buf = 0;

    for (int kk = 0; kk < K; kk += 32) {
        if (kk + 32 < K) stage(buf ^ 1, kk + 32);  // async, overlaps WMMAs below
        if (kk + 64 < K) {   // lowers to global_prefetch_b8
            __builtin_prefetch(A + (size_t)(m0 + sra) * lda + kk + 64 + ska, 0, 1);
            __builtin_prefetch(Bt + (size_t)(n0 + tid) * ldb + kk + 64, 0, 1);
        }

        Frag16 fb[4], fa[4];
        for (int j = 0; j < 4; j++) {
            const u16* p = &Bs[buf][(wn * 64 + j * 16 + (lane & 15)) * LSTR + bk0];
            fb[j].h[0] = *(const v8bf*)(p);
            fb[j].h[1] = *(const v8bf*)(p + 16);
        }
        for (int a = 0; a < 4; a++) {
            const u16* p = &As[buf][(wm * 64 + a * 16 + (lane & 15)) * LSTR];
            fa[a].h[0] = *(const v8bf*)(p + ak0);
            fa[a].h[1] = *(const v8bf*)(p + ak0 + 16);
        }
        for (int a = 0; a < 4; a++)
            for (int j = 0; j < 4; j++)
                acc[a][j] = wmma_bf16(fa[a].v, fb[j].v, acc[a][j]);

        wait_async0();       // this wave's copies into buf^1 have landed
        __syncthreads();     // everyone's copies visible; buf reads done
        buf ^= 1;
    }

    // epilogue (C layout: lane holds N=lane%16, rows v + 8*(lane>=16))
    const int rowoff = (lane & 16) ? 8 : 0;
    for (int a = 0; a < 4; a++) {
        for (int j = 0; j < 4; j++) {
            for (int v = 0; v < 8; v++) {
                int m = m0 + wm * 64 + a * 16 + v + rowoff;
                int n = n0 + wn * 64 + j * 16 + (lane & 15);
                if (m < M && n < N) {
                    float val = acc[a][j][v];
                    if (bias)  val += bias[n];
                    if (relu)  val = fmaxf(val, 0.0f);
                    if (resid) val += resid[(size_t)m * ldr + n];
                    if (out_bf16) ((u16*)out)[(size_t)m * ldo + n] = f2bf(val);
                    else          ((float*)out)[(size_t)m * ldo + n] = val;
                }
            }
        }
    }
}

// ---------------------------------------------------------------------------
// Flash-style attention per (b,h): Q,K,V bf16 packed as [B*T][3*D] (ld=3072),
// output O bf16 [B*T][D].  128 threads / 4 waves; wave owns 16 q-rows.
// ---------------------------------------------------------------------------
static constexpr int PSTR = 80;   // LDS stride keeping all chunk loads 16B aligned

__global__ __launch_bounds__(128) void attn_kernel(
    const u16* __restrict__ Qg, const u16* __restrict__ Kg,
    const u16* __restrict__ Vg, int ld, u16* __restrict__ Og)
{
    __shared__ u16 Vt[64 * PSTR];        // V tile transposed: [d][s]
    __shared__ u16 Pl[4][16 * PSTR];     // per-wave P tile: [qrow][s]

    const int tid  = threadIdx.x;
    const int lane = tid & 31;
    const int wave = tid >> 5;
    const int bh = blockIdx.y;
    const int bb = bh / NHEAD, hh = bh - bb * NHEAD;
    const int q0 = blockIdx.x * 64 + wave * 16;     // q-row base for this wave
    const size_t base  = (size_t)bb * SEQ * ld + (size_t)hh * HEADSZ;
    const size_t obase = (size_t)bb * SEQ * DIM + (size_t)hh * HEADSZ;

    const int arow = lane & 15;
    const int ak0  = (lane & 16) ? 8 : 0;
    const int bk0  = (lane & 16) ? 8 : 0;

    // Q fragments (A layout), held in registers for all key blocks
    Frag16 fq[2];
    for (int kt = 0; kt < 2; kt++) {
        const u16* p = Qg + base + (size_t)(q0 + arow) * ld + kt * 32;
        fq[kt].h[0] = *(const v8bf*)(p + ak0);
        fq[kt].h[1] = *(const v8bf*)(p + ak0 + 16);
    }

    v8f o[4] = {};
    float rm[8], rl[8];
    for (int v = 0; v < 8; v++) { rm[v] = -1e30f; rl[v] = 0.f; }

    const float scale = 0.125f;   // HS^-0.5

    for (int sb = 0; sb < SEQ / 64; sb++) {
        const int s0 = sb * 64;

        __syncthreads();   // previous block's Vt reads complete
        // stage V tile transposed (cooperative, 64x64)
        for (int i = tid; i < 64 * 64; i += 128) {
            int s = i >> 6, d = i & 63;
            Vt[d * PSTR + s] = Vg[base + (size_t)(s0 + s) * ld + d];
        }

        // S = Q * K^T  (K rows [s][d] are exactly the B^T fragment layout)
        v8f sacc[4] = {};
        for (int j = 0; j < 4; j++) {
            for (int kt = 0; kt < 2; kt++) {
                const u16* p = Kg + base +
                    (size_t)(s0 + j * 16 + (lane & 15)) * ld + kt * 32 + bk0;
                Frag16 fb;
                fb.h[0] = *(const v8bf*)(p);
                fb.h[1] = *(const v8bf*)(p + 16);
                sacc[j] = wmma_bf16(fq[kt].v, fb.v, sacc[j]);
            }
        }
        __syncthreads();   // Vt staging visible

        // online softmax (row = v + 8*(lane>=16); 16-lane row reductions)
        for (int v = 0; v < 8; v++) {
            float p0 = sacc[0][v] * scale, p1 = sacc[1][v] * scale;
            float p2 = sacc[2][v] * scale, p3 = sacc[3][v] * scale;
            float bm = fmaxf(fmaxf(p0, p1), fmaxf(p2, p3));
            bm = fmaxf(bm, __shfl_xor(bm, 8));
            bm = fmaxf(bm, __shfl_xor(bm, 4));
            bm = fmaxf(bm, __shfl_xor(bm, 2));
            bm = fmaxf(bm, __shfl_xor(bm, 1));
            float newm = fmaxf(rm[v], bm);
            float corr = __expf(rm[v] - newm);
            rm[v] = newm;
            p0 = __expf(p0 - newm); p1 = __expf(p1 - newm);
            p2 = __expf(p2 - newm); p3 = __expf(p3 - newm);
            float bs = p0 + p1 + p2 + p3;
            bs += __shfl_xor(bs, 8);
            bs += __shfl_xor(bs, 4);
            bs += __shfl_xor(bs, 2);
            bs += __shfl_xor(bs, 1);
            rl[v] = rl[v] * corr + bs;
            for (int j2 = 0; j2 < 4; j2++) o[j2][v] *= corr;
            // stash P (C layout -> LDS row-major) for A-fragment reload
            const int r = v + ((lane & 16) ? 8 : 0);
            u16* pb = &Pl[wave][r * PSTR + (lane & 15)];
            pb[0]  = f2bf(p0);
            pb[16] = f2bf(p1);
            pb[32] = f2bf(p2);
            pb[48] = f2bf(p3);
        }

        // O += P * V   (A from Pl, B from transposed Vt)
        for (int kt = 0; kt < 2; kt++) {
            Frag16 fp;
            const u16* pp = &Pl[wave][arow * PSTR + kt * 32 + ak0];
            fp.h[0] = *(const v8bf*)(pp);
            fp.h[1] = *(const v8bf*)(pp + 16);
            for (int j2 = 0; j2 < 4; j2++) {
                const u16* pv = &Vt[(j2 * 16 + (lane & 15)) * PSTR + kt * 32 + bk0];
                Frag16 fv;
                fv.h[0] = *(const v8bf*)(pv);
                fv.h[1] = *(const v8bf*)(pv + 16);
                o[j2] = wmma_bf16(fp.v, fv.v, o[j2]);
            }
        }
    }

    // normalize and write O (bf16, heads concatenated along columns)
    for (int v = 0; v < 8; v++) {
        const float inv = 1.0f / rl[v];
        const int m = q0 + v + ((lane & 16) ? 8 : 0);
        for (int j2 = 0; j2 < 4; j2++)
            Og[obase + (size_t)m * DIM + j2 * 16 + (lane & 15)] = f2bf(o[j2][v] * inv);
    }
}

// ---------------------------------------------------------------------------
// Host-side orchestration
// ---------------------------------------------------------------------------
extern "C" void kernel_launch(void* const* d_in, const int* in_sizes, int n_in,
                              void* d_out, int out_size, void* d_ws, size_t ws_size,
                              hipStream_t stream) {
    const float* x     = (const float*)d_in[0];
    const float* Wq    = (const float*)d_in[1];
    const float* Wk    = (const float*)d_in[2];
    const float* Wv    = (const float*)d_in[3];
    const float* Wo    = (const float*)d_in[4];
    const float* bo    = (const float*)d_in[5];
    const float* W1    = (const float*)d_in[6];
    const float* b1    = (const float*)d_in[7];
    const float* W2    = (const float*)d_in[8];
    const float* b2    = (const float*)d_in[9];
    const float* ln1g  = (const float*)d_in[10];
    const float* ln1b  = (const float*)d_in[11];
    const float* ln2g  = (const float*)d_in[12];
    const float* ln2b  = (const float*)d_in[13];
    float* outp = (float*)d_out;

    char* ws = (char*)d_ws;
    size_t off = 0;
    auto alloc = [&](size_t bytes) -> void* {
        void* p = ws + off;
        off += (bytes + 255) & ~(size_t)255;
        return p;
    };
    u16* hA   = (u16*)alloc((size_t)MROWS * DIM * 2);     // LN output (reused for ln2)
    u16* WqT  = (u16*)alloc((size_t)DIM * DIM * 2);       // WqT/WkT/WvT contiguous =>
    u16* WkT  = (u16*)alloc((size_t)DIM * DIM * 2);       //   one fused [3072][1024] B
    u16* WvT  = (u16*)alloc((size_t)DIM * DIM * 2);
    u16* WoT  = (u16*)alloc((size_t)DIM * DIM * 2);
    u16* W1T  = (u16*)alloc((size_t)FFDIM * DIM * 2);
    u16* W2T  = (u16*)alloc((size_t)DIM * FFDIM * 2);
    u16* QKVb = (u16*)alloc((size_t)MROWS * QKVN * 2);    // fused Q|K|V, ld=3072
    u16* Ob   = (u16*)alloc((size_t)MROWS * DIM * 2);
    u16* A1   = (u16*)alloc((size_t)MROWS * FFDIM * 2);

    // --- weight repacks (fp32 -> bf16, K-contiguous transposed) ---
    const int n_dd = (DIM * DIM + 255) / 256;
    repack_qkv_kernel<<<n_dd, 256, 0, stream>>>(Wq, WqT);
    repack_qkv_kernel<<<n_dd, 256, 0, stream>>>(Wk, WkT);
    repack_qkv_kernel<<<n_dd, 256, 0, stream>>>(Wv, WvT);
    transpose_conv_kernel<<<n_dd, 256, 0, stream>>>(Wo, WoT, DIM, DIM);
    const int n_df = (DIM * FFDIM + 255) / 256;
    transpose_conv_kernel<<<n_df, 256, 0, stream>>>(W1, W1T, DIM, FFDIM);
    transpose_conv_kernel<<<n_df, 256, 0, stream>>>(W2, W2T, FFDIM, DIM);

    // --- h = LN1(x) ---
    ln_kernel<<<MROWS, 256, 0, stream>>>(x, ln1g, ln1b, hA);

    // --- fused QKV projection: [4096x1024] x [3072x1024]^T -> [4096x3072] ---
    dim3 gqkv(QKVN / 256, MROWS / 128);
    gemm_bf16_kernel<<<gqkv, 256, 0, stream>>>(hA, DIM, WqT, DIM, MROWS, QKVN, DIM,
                                               nullptr, nullptr, 0, QKVb, QKVN, 1, 0);

    // --- attention ---
    dim3 gattn(SEQ / 64, BATCH * NHEAD);
    attn_kernel<<<gattn, 128, 0, stream>>>(QKVb, QKVb + DIM, QKVb + 2 * DIM, QKVN, Ob);

    // --- out-projection + residual: x_mid = x + O*Wo + bo  (fp32 in d_out) ---
    dim3 gproj(DIM / 256, MROWS / 128);
    gemm_bf16_kernel<<<gproj, 256, 0, stream>>>(Ob, DIM, WoT, DIM, MROWS, DIM, DIM,
                                                bo, x, DIM, outp, DIM, 0, 0);

    // --- h2 = LN2(x_mid) ---
    ln_kernel<<<MROWS, 256, 0, stream>>>(outp, ln2g, ln2b, hA);

    // --- FFN1: relu(h2*W1 + b1) -> bf16 ---
    dim3 gff1(FFDIM / 256, MROWS / 128);
    gemm_bf16_kernel<<<gff1, 256, 0, stream>>>(hA, DIM, W1T, DIM, MROWS, FFDIM, DIM,
                                               b1, nullptr, 0, A1, FFDIM, 1, 1);

    // --- FFN2: out = x_mid + a1*W2 + b2 (fp32; resid==out, per-element RAW only) ---
    gemm_bf16_kernel<<<gproj, 256, 0, stream>>>(A1, FFDIM, W2T, FFDIM, MROWS, DIM, FFDIM,
                                                b2, outp, DIM, outp, DIM, 0, 0);

    (void)in_sizes; (void)n_in; (void)out_size; (void)ws_size;
}
